// ScaledDotProductAttention_28389733826852
// MI455X (gfx1250) — matrix operand order repaired and checked by reading, not compiled, not run on hardware
//
#include <hip/hip_runtime.h>
#include <hip/hip_bf16.h>
#include <math.h>

// ---------------------------------------------------------------------------
// ScaledDotProductAttention for MI455X (gfx1250), wave32, bf16 WMMA + f32 acc.
// B=4, S=4096, D_MODEL=1024, D_K=D_V=64.
// ---------------------------------------------------------------------------

#define B_    4
#define S_    4096
#define DM_   1024
#define DK_   64
#define ROWS_ (B_ * S_)          // 16384 token rows
#define CH_   64                 // keys per attention chunk
// log2(e) / sqrt(d_k): folds softmax exp->exp2 conversion into Q projection.
#define QSCALE_ 0.18033688011112042f

typedef __attribute__((ext_vector_type(16))) __bf16 v16bf;
typedef __attribute__((ext_vector_type(8)))  float  v8f;

// Plain-POD 128/256-bit carriers (HIP's uint4/float4 have ctors -> not union-safe).
struct alignas(16) U128 { unsigned x, y, z, w; };
struct alignas(16) F128 { float x, y, z, w; };
struct U256 { U128 a, b; };

static __device__ inline v16bf ld16(const void* p0, const void* p1) {
  U256 t;
  t.a = *reinterpret_cast<const U128*>(p0);
  t.b = *reinterpret_cast<const U128*>(p1);
  return __builtin_bit_cast(v16bf, t);
}

static __device__ inline unsigned short bf16bits(float x) {
  __bf16 h = (__bf16)x;
  unsigned short u;
  __builtin_memcpy(&u, &h, 2);
  return u;
}

// ---------------------------------------------------------------------------
// Kernel 1: QKV projection (unchanged from round 1 -- codegen verified good).
//   Qb[row][n] = ((emb @ Wq)[row][n] + bq[n]) * QSCALE      (bf16)
//   Kb[row][n] =  (emb @ Wk)[row][n] + bk[n]                (bf16)
//   Vt[b][n][s] = (emb @ Wv)[row][n] + bv[n]                (bf16, transposed)
// ---------------------------------------------------------------------------
__global__ __launch_bounds__(256) void qkv_project_kernel(
    const float* __restrict__ emb,
    const float* __restrict__ Wq, const float* __restrict__ bq,
    const float* __restrict__ Wk, const float* __restrict__ bk,
    const float* __restrict__ Wv, const float* __restrict__ bv,
    __bf16* __restrict__ Qb, __bf16* __restrict__ Kb, __bf16* __restrict__ Vt) {

  // [matrix][n=64][d=32 (+8 pad)] bf16; row stride 80B -> conflict-free, 16B aligned.
  __shared__ __align__(16) __bf16 ldsW[3][64][40];

  const int t    = threadIdx.x;
  const int wave = t >> 5;
  const int lane = t & 31;
  const int ln   = lane & 15;
  const int hi   = lane >> 4;
  const int rowbase = blockIdx.x * 128 + wave * 16;

  const float* Ws[3] = {Wq, Wk, Wv};

  const v8f z = {0.f, 0.f, 0.f, 0.f, 0.f, 0.f, 0.f, 0.f};
  v8f acc[3][4];
#pragma unroll
  for (int m = 0; m < 3; ++m)
#pragma unroll
    for (int nt = 0; nt < 4; ++nt) acc[m][nt] = z;

  for (int kc = 0; kc < DM_ / 32; ++kc) {
    __syncthreads();  // protect previous iteration's LDS reads
#pragma unroll
    for (int m = 0; m < 3; ++m) {
#pragma unroll
      for (int i = 0; i < 8; ++i) {
        int flat = t + 256 * i;            // 0..2047
        int d = flat >> 6, n = flat & 63;  // coalesced over n
        ldsW[m][n][d] = (__bf16)Ws[m][(size_t)(kc * 32 + d) * DK_ + n];
      }
    }
    __syncthreads();

    // A operand: 16x32 tile of embeddings, f32 -> bf16, ISA A-layout.
    const float* ap = emb + (size_t)(rowbase + ln) * DM_ + kc * 32 + 8 * hi;
    F128 f0 = *(const F128*)(ap);
    F128 f1 = *(const F128*)(ap + 4);
    F128 f2 = *(const F128*)(ap + 16);
    F128 f3 = *(const F128*)(ap + 20);
    float fa[16] = {f0.x, f0.y, f0.z, f0.w, f1.x, f1.y, f1.z, f1.w,
                    f2.x, f2.y, f2.z, f2.w, f3.x, f3.y, f3.z, f3.w};
    v16bf av;
#pragma unroll
    for (int i = 0; i < 16; ++i) av[i] = (__bf16)fa[i];

#pragma unroll
    for (int m = 0; m < 3; ++m) {
#pragma unroll
      for (int nt = 0; nt < 4; ++nt) {
        const __bf16* bp = &ldsW[m][nt * 16 + ln][8 * hi];
        v16bf bv_ = ld16(bp, bp + 16);
        acc[m][nt] = __builtin_amdgcn_wmma_f32_16x16x32_bf16(
            false, av, false, bv_, (short)0, acc[m][nt], false, false);
      }
    }
  }

  // Epilogue: bias, Q scale, stores (V transposed & packed 8xbf16 = 16B).
  const int bidx = rowbase / S_;
  const int srow = (rowbase % S_) + 8 * hi;
#pragma unroll
  for (int nt = 0; nt < 4; ++nt) {
    const int n = nt * 16 + ln;
    const float bqv = bq[n], bkv = bk[n], bvv = bv[n];
#pragma unroll
    for (int r = 0; r < 8; ++r) {
      const size_t row = (size_t)(rowbase + 8 * hi + r);
      Qb[row * DK_ + n] = (__bf16)((acc[0][nt][r] + bqv) * QSCALE_);
      Kb[row * DK_ + n] = (__bf16)(acc[1][nt][r] + bkv);
    }
    unsigned vp[4];
#pragma unroll
    for (int rp = 0; rp < 4; ++rp) {
      float v0 = acc[2][nt][2 * rp] + bvv;
      float v1 = acc[2][nt][2 * rp + 1] + bvv;
      vp[rp] = (unsigned)bf16bits(v0) | ((unsigned)bf16bits(v1) << 16);
    }
    U128 pk = {vp[0], vp[1], vp[2], vp[3]};
    *reinterpret_cast<U128*>(Vt + ((size_t)bidx * DK_ + n) * S_ + srow) = pk;
  }
}

// ---------------------------------------------------------------------------
// Kernel 2: block-cooperative flash attention.
// 256 threads = 8 waves = 128 q rows per block. K / V^T chunks (64 keys) are
// staged into double-buffered LDS once per block (8x less L2 traffic than
// per-wave streaming), with global->register prefetch overlapping the WMMAs.
// Scores computed transposed (S^T = K_tile @ Q^T) so the softmax row lives in
// a lane: row reduction = 1 shfl_xor. Online softmax in exp2 domain (scale
// folded into Q). O^T = V^T @ P^T accumulated in f32.
// ---------------------------------------------------------------------------
__global__ __launch_bounds__(256) void attn_kernel(
    const __bf16* __restrict__ Qb, const __bf16* __restrict__ Kb,
    const __bf16* __restrict__ Vt, float* __restrict__ out) {

  // Row stride 72 bf16 = 144B: 16B-aligned rows, conflict-free WMMA-operand reads.
  __shared__ __align__(16) __bf16 ldsK[2][CH_][72];  // [buf][kpos][d]
  __shared__ __align__(16) __bf16 ldsV[2][CH_][72];  // [buf][dv][s]
  __shared__ __align__(16) __bf16 ldsP[8][16][72];   // per-wave P^T staging [q][kpos]

  const int t    = threadIdx.x;
  const int wave = t >> 5;
  const int lane = t & 31;
  const int ln   = lane & 15;
  const int hi   = lane >> 4;

  const int b     = blockIdx.x >> 5;                    // 32 blocks per batch
  const int qbase = (blockIdx.x & 31) * 128 + wave * 16;
  const size_t qrow = (size_t)b * S_ + qbase + ln;

  // Q^T B-operands, loaded once (d chunks 0-31, 32-63).
  v16bf qB[2];
#pragma unroll
  for (int c = 0; c < 2; ++c) {
    const __bf16* qp = Qb + qrow * DK_ + 32 * c + 8 * hi;
    qB[c] = ld16(qp, qp + 16);
  }

  // Cooperative staging role: thread -> (row 0..63, 32B part 0..3). Coalesced.
  const int srow  = t >> 2;
  const int spart = (t & 3) * 16;  // in bf16 elements (32B)
  U128 kreg0, kreg1, vreg0, vreg1;

  auto stage_load = [&](int kb) {
    const __bf16* kp = Kb + (size_t)(b * S_ + kb + srow) * DK_ + spart;
    kreg0 = *reinterpret_cast<const U128*>(kp);
    kreg1 = *reinterpret_cast<const U128*>(kp + 8);
    const __bf16* vp = Vt + ((size_t)b * DK_ + srow) * S_ + kb + spart;
    vreg0 = *reinterpret_cast<const U128*>(vp);
    vreg1 = *reinterpret_cast<const U128*>(vp + 8);
  };
  auto stage_store = [&](int bf) {
    __bf16* kd = &ldsK[bf][srow][spart];
    *reinterpret_cast<U128*>(kd) = kreg0;
    *reinterpret_cast<U128*>(kd + 8) = kreg1;
    __bf16* vd = &ldsV[bf][srow][spart];
    *reinterpret_cast<U128*>(vd) = vreg0;
    *reinterpret_cast<U128*>(vd + 8) = vreg1;
  };

  const v8f z = {0.f, 0.f, 0.f, 0.f, 0.f, 0.f, 0.f, 0.f};
  v8f ot[4] = {z, z, z, z};           // O^T accumulators, 4 dv-tiles
  float m = -INFINITY, lsum = 0.f;

  // Prologue: stage chunk 0.
  stage_load(0);
  stage_store(0);
  __syncthreads();

  int buf = 0;
  for (int kb = 0; kb < S_; kb += CH_) {
    const bool has_next = (kb + CH_) < S_;
    if (has_next) stage_load(kb + CH_);   // overlap global latency with WMMAs

    // ---- scores^T: four 16(kpos)x16(q) tiles, contraction over d=64 ----
    v8f s[4];
#pragma unroll
    for (int t2 = 0; t2 < 4; ++t2) {
      v8f c = z;
#pragma unroll
      for (int c2 = 0; c2 < 2; ++c2) {
        const __bf16* kp = &ldsK[buf][16 * t2 + ln][32 * c2 + 8 * hi];
        v16bf a = ld16(kp, kp + 16);
        c = __builtin_amdgcn_wmma_f32_16x16x32_bf16(
            false, a, false, qB[c2], (short)0, c, false, false);
      }
      s[t2] = c;
    }

    // ---- online softmax over 64 keys (exp2 domain) ----
    float mp = s[0][0];
#pragma unroll
    for (int t2 = 0; t2 < 4; ++t2)
#pragma unroll
      for (int r = 0; r < 8; ++r) mp = fmaxf(mp, s[t2][r]);
    mp = fmaxf(mp, __shfl_xor(mp, 16, 32));   // combine kpos half-rows
    const float mnew = fmaxf(m, mp);
    const float corr = exp2f(m - mnew);       // first iter: exp2(-inf)=0
    m = mnew;
    lsum *= corr;
#pragma unroll
    for (int dt = 0; dt < 4; ++dt)
#pragma unroll
      for (int r = 0; r < 8; ++r) ot[dt][r] *= corr;

    // p = exp2(s - m), accumulate l, pack pairs -> per-wave LDS (P^T B-layout).
#pragma unroll
    for (int t2 = 0; t2 < 4; ++t2) {
#pragma unroll
      for (int rp = 0; rp < 4; ++rp) {
        float p0 = exp2f(s[t2][2 * rp] - mnew);
        float p1 = exp2f(s[t2][2 * rp + 1] - mnew);
        lsum += p0 + p1;
        unsigned u = (unsigned)bf16bits(p0) | ((unsigned)bf16bits(p1) << 16);
        *reinterpret_cast<unsigned*>(
            &ldsP[wave][ln][16 * t2 + 8 * hi + 2 * rp]) = u;
      }
    }
    asm volatile("s_wait_dscnt 0" ::: "memory");  // wave-private RAW on LDS

    // ---- O^T += V^T @ P^T : 2 kpos halves x 4 dv-tiles = 8 WMMAs ----
#pragma unroll
    for (int kc2 = 0; kc2 < 2; ++kc2) {
      const __bf16* pp = &ldsP[wave][ln][32 * kc2 + 8 * hi];
      v16bf pB = ld16(pp, pp + 16);
#pragma unroll
      for (int dt = 0; dt < 4; ++dt) {
        const __bf16* vp = &ldsV[buf][16 * dt + ln][32 * kc2 + 8 * hi];
        v16bf a = ld16(vp, vp + 16);
        ot[dt] = __builtin_amdgcn_wmma_f32_16x16x32_bf16(
            false, a, false, pB, (short)0, ot[dt], false, false);
      }
    }

    // Double buffer: stores go to buf^1 while (this iter) reads hit buf;
    // single barrier per chunk orders store(i)->read(i+1) and read(i)->store(i+1).
    if (has_next) stage_store(buf ^ 1);
    __syncthreads();
    buf ^= 1;
  }

  // ---- normalize and store O (f32, coalesced 16B stores) ----
  lsum += __shfl_xor(lsum, 16, 32);
  const float inv = 1.0f / lsum;
#pragma unroll
  for (int dt = 0; dt < 4; ++dt) {
    float* op = out + qrow * DK_ + 16 * dt + 8 * hi;
    F128 lo = {ot[dt][0] * inv, ot[dt][1] * inv, ot[dt][2] * inv, ot[dt][3] * inv};
    F128 hi4 = {ot[dt][4] * inv, ot[dt][5] * inv, ot[dt][6] * inv, ot[dt][7] * inv};
    *reinterpret_cast<F128*>(op) = lo;
    *reinterpret_cast<F128*>(op + 4) = hi4;
  }
}

// ---------------------------------------------------------------------------
// Launch: ws holds bf16 Q (2MB) | K (2MB) | V^T (2MB).
// ---------------------------------------------------------------------------
extern "C" void kernel_launch(void* const* d_in, const int* in_sizes, int n_in,
                              void* d_out, int out_size, void* d_ws, size_t ws_size,
                              hipStream_t stream) {
  const float* emb = (const float*)d_in[0];
  const float* Wq  = (const float*)d_in[1];
  const float* bq  = (const float*)d_in[2];
  const float* Wk  = (const float*)d_in[3];
  const float* bk  = (const float*)d_in[4];
  const float* Wv  = (const float*)d_in[5];
  const float* bv  = (const float*)d_in[6];
  float* out = (float*)d_out;

  const size_t elems = (size_t)ROWS_ * DK_;       // 1,048,576 bf16 each
  __bf16* Qb = (__bf16*)d_ws;
  __bf16* Kb = Qb + elems;
  __bf16* Vt = Kb + elems;

  qkv_project_kernel<<<dim3(ROWS_ / 128), dim3(256), 0, stream>>>(
      emb, Wq, bq, Wk, bk, Wv, bv, Qb, Kb, Vt);
  attn_kernel<<<dim3(ROWS_ / 128), dim3(256), 0, stream>>>(
      Qb, Kb, Vt, out);
}
